// DGCNNEncoder_59253368815618
// MI455X (gfx1250) — matrix operand order repaired and checked by reading, not compiled, not run on hardware
//
#include <hip/hip_runtime.h>
#include <cstdint>
#include <cstddef>

// ---------------- problem constants (from reference setup) ----------------
#define BATCH 8
#define NPTS  2048
#define KNN   20
#define NKCOL (NPTS * KNN)        // 40960 gather columns per batch
#define EPSBN 1e-5f

typedef _Float16 v16h __attribute__((ext_vector_type(16)));
typedef _Float16 v8h  __attribute__((ext_vector_type(8)));
typedef float    v8f  __attribute__((ext_vector_type(8)));

__device__ __forceinline__ void atomicMaxPosF(float* p, float v) {
    // valid for non-negative floats: bit pattern order == value order
    atomicMax((unsigned int*)p, __float_as_uint(v));
}
__device__ __forceinline__ void atomicMaxPosF_lds(float* p, float v) {
    atomicMax((unsigned int*)p, __float_as_uint(v));
}

// ---------------- utility kernels ----------------
__global__ void zero_f32(float* __restrict__ p, int n) {
    int i = blockIdx.x * 256 + threadIdx.x;
    if (i < n) p[i] = 0.0f;
}

// points (B,N,3) -> x0 (B,3,N)
__global__ void pts_to_cxn(const float* __restrict__ pts, float* __restrict__ x0) {
    int i = blockIdx.x * 256 + threadIdx.x;
    if (i >= BATCH * 3 * NPTS) return;
    int n = i % NPTS;
    int c = (i / NPTS) % 3;
    int b = i / (3 * NPTS);
    x0[i] = pts[(b * NPTS + n) * 3 + c];
}

// fp32 weights (O,Kin) -> fp16 (O,Kpad), zero padded along K
__global__ void w_to_f16(const float* __restrict__ w, _Float16* __restrict__ wh,
                         int O, int Kin, int Kpad) {
    int i = blockIdx.x * 256 + threadIdx.x;
    if (i >= O * Kpad) return;
    int o = i / Kpad, c = i % Kpad;
    wh[i] = (c < Kin) ? (_Float16)w[o * Kin + c] : (_Float16)0.0f;
}

// ---------------- KNN (top-20 of pairwise -||xi-xj||^2) ----------------
// One thread per query point n; LDS tile of 64 candidate columns.
template <int C>
__global__ __launch_bounds__(128) void knn_kernel(const float* __restrict__ x,
                                                  int bstride,
                                                  int* __restrict__ idxout) {
    __shared__ float sM[C * 64];
    __shared__ float sXX[64];
    const int tid = threadIdx.x;
    const int b = blockIdx.y;
    const int n = blockIdx.x * 128 + tid;
    const float* xb = x + (size_t)b * bstride;

    float xn[C];
    float xxn = 0.0f;
#pragma unroll
    for (int c = 0; c < C; ++c) {
        xn[c] = xb[c * NPTS + n];
        xxn += xn[c] * xn[c];
    }

    float bd[KNN];
    int   bi[KNN];
#pragma unroll
    for (int t = 0; t < KNN; ++t) { bd[t] = -3.0e38f; bi[t] = 0; }

    for (int mt = 0; mt < NPTS / 64; ++mt) {
        __syncthreads();
        for (int e = tid; e < C * 64; e += 128) {
            int c = e >> 6, col = e & 63;
            sM[e] = xb[c * NPTS + mt * 64 + col];
        }
        __syncthreads();
        if (tid < 64) {
            float s = 0.0f;
#pragma unroll
            for (int c = 0; c < C; ++c) {
                float v = sM[c * 64 + tid];
                s += v * v;
            }
            sXX[tid] = s;
        }
        __syncthreads();
        for (int mm = 0; mm < 64; ++mm) {
            float dot = 0.0f;
#pragma unroll
            for (int c = 0; c < C; ++c) dot += xn[c] * sM[c * 64 + mm];
            float d = 2.0f * dot - xxn - sXX[mm];
            int m = mt * 64 + mm;
            if (d > bd[KNN - 1]) {
                int j = KNN - 1;
#pragma unroll
                for (int t = KNN - 1; t > 0; --t) {
                    if (d > bd[t - 1]) {            // sorted descending -> predicated shift
                        bd[t] = bd[t - 1];
                        bi[t] = bi[t - 1];
                        j = t - 1;
                    }
                }
                bd[j] = d;
                bi[j] = m;
            }
        }
    }
#pragma unroll
    for (int t = 0; t < KNN; ++t) idxout[((size_t)b * NPTS + n) * KNN + t] = bi[t];
}

// ---------------- fused edge-conv / projection GEMM (WMMA f16->f32) ----------------
// GATHER=true : 80-column panels = exactly 4 points x 20 neighbors. Max over k is
//               reduced in LDS and plain-stored (no global atomics, no init needed).
//               f[c] = x[c][nbr]-x[c][n] (c<Cin) | x[c-Cin][n] (center). Kpad<=256.
// GATHER=false: 32-column panels of n; block-level max over n in LDS, then one
//               global atomic-max per channel into xg. Kpad<=512.
// PASS=1: accumulate per-channel sum / sumsq of h (for BN), LDS-first then flush.
// PASS=2: h*scale+shift, ReLU, max-reduce as above.
template <bool GATHER, int PASS>
__global__ __launch_bounds__(256) void gemm_conv(
    const float* __restrict__ xin, int in_bstride,
    const int* __restrict__ idx,
    const _Float16* __restrict__ wh,
    int O, int Cin, int Kpad,
    float* __restrict__ stats,
    const float* __restrict__ scsh,
    float* __restrict__ outp, int out_bstride, int out_choff) {

    constexpr int PCOLS  = GATHER ? 80 : 32;           // columns per panel
    constexpr int CTILES = PCOLS / 16;                 // 16-wide column tiles
    constexpr int SFELEM = GATHER ? (80 * 272) : (32 * 528);

    __shared__ __align__(32) _Float16 sF[SFELEM];      // [PCOLS][Kpad+16]
    __shared__ float sStat[2048];                      // [sum(0..1023) | sumsq(1024..2047)]
    __shared__ float sRed[1024];                       // k/n max reduction (pass 2)

    const int tid = threadIdx.x;
    const int b = blockIdx.y;
    const int panel = blockIdx.x;
    const int stride = Kpad + 16;                      // halves; mult of 16 -> 32B-aligned frags
    const float* xb = xin + (size_t)b * in_bstride;

    if (PASS == 1) {
        for (int i = tid; i < 2048; i += 256) sStat[i] = 0.0f;
    } else {
        for (int i = tid; i < 1024; i += 256) sRed[i] = 0.0f;  // ReLU floor
    }

    // stage fp16 feature panel into LDS (gather + edge feature on the fly)
    for (int e = tid; e < Kpad * PCOLS; e += 256) {
        int c = e / PCOLS, col = e % PCOLS;
        int j = panel * PCOLS + col;
        float v;
        if (GATHER) {
            int n = j / KNN, kk = j % KNN;
            if (c < Cin) {
                int nb = idx[((size_t)b * NPTS + n) * KNN + kk];
                v = xb[c * NPTS + nb] - xb[c * NPTS + n];
            } else if (c < 2 * Cin) {
                v = xb[(c - Cin) * NPTS + n];
            } else {
                v = 0.0f;
            }
        } else {
            v = (c < Cin) ? xb[c * NPTS + j] : 0.0f;
        }
        sF[col * stride + c] = (_Float16)v;
    }
    __syncthreads();

    const int wave = tid >> 5, lane = tid & 31;
    const int laneN = lane & 15, laneHi = lane >> 4;
    const int numTiles = (O / 16) * CTILES;

    for (int t = wave; t < numTiles; t += 8) {
        const int otile = t / CTILES, ctile = t % CTILES;
        const int col = ctile * 16 + laneN;
        const int orow = otile * 16 + laneN;
        // prefetch next A tile (global_prefetch_b8)
        __builtin_prefetch(wh + (size_t)orow * Kpad, 0, 1);
        v8f acc = {};
        for (int ck = 0; ck < Kpad; ck += 32) {
            // B fragment: 16 consecutive K per lane, upper half-wave offset by 16
            v16h bf = *(const v16h*)(&sF[col * stride + ck + laneHi * 16]);
            // A fragment: K = ck + laneHi*16 + {0..15}
            const _Float16* ap = wh + (size_t)orow * Kpad + ck + laneHi * 16;
            v16h af = *(const v16h*)(ap);
            acc = __builtin_amdgcn_wmma_f32_16x16x32_f16(
                false, af, false, bf, (short)0, acc, false, false);
        }
        if (PASS == 1) {
#pragma unroll
            for (int e = 0; e < 8; ++e) {
                int o = otile * 16 + e + laneHi * 8;
                float v = acc[e];
                atomicAdd(&sStat[o], v);
                atomicAdd(&sStat[1024 + o], v * v);
            }
        } else {
#pragma unroll
            for (int e = 0; e < 8; ++e) {
                int o = otile * 16 + e + laneHi * 8;
                float y = acc[e] * scsh[o] + scsh[O + o];
                y = fmaxf(y, 0.0f);
                if (GATHER) {
                    int nl = col / KNN;                // 0..3: point within panel
                    atomicMaxPosF_lds(&sRed[o * 4 + nl], y);
                } else {
                    atomicMaxPosF_lds(&sRed[o], y);
                }
            }
        }
    }
    __syncthreads();

    if (PASS == 1) {
        for (int o = tid; o < O; o += 256) {
            atomicAdd(&stats[o], sStat[o]);
            atomicAdd(&stats[O + o], sStat[1024 + o]);
        }
    } else if (GATHER) {
        // block owns all 20 neighbor columns of its 4 points -> plain store
        for (int i = tid; i < O * 4; i += 256) {
            int o = i >> 2, nl = i & 3;
            int n = panel * 4 + nl;
            outp[(size_t)b * out_bstride + (size_t)(out_choff + o) * NPTS + n] = sRed[i];
        }
    } else {
        // global max pool: one atomic per channel per block
        for (int o = tid; o < O; o += 256) {
            atomicMaxPosF(&outp[b * O + o], sRed[o]);
        }
    }
}

// sums -> BN scale/shift
__global__ void bn_finalize(const float* __restrict__ stats,
                            const float* __restrict__ g, const float* __restrict__ bb,
                            float* __restrict__ scsh, int O, float invcnt) {
    int o = blockIdx.x * 256 + threadIdx.x;
    if (o >= O) return;
    float mean = stats[o] * invcnt;
    float var = stats[O + o] * invcnt - mean * mean;
    float sc = g[o] * rsqrtf(var + EPSBN);
    scsh[o] = sc;
    scsh[O + o] = bb[o] - mean * sc;
}

// out[b,o] = relu(xg[b] . fcw[o] + fcb[o])
__global__ void fc_relu(const float* __restrict__ xg, const float* __restrict__ fcw,
                        const float* __restrict__ fcb, float* __restrict__ out) {
    int i = blockIdx.x * 256 + threadIdx.x;
    if (i >= BATCH * 256) return;
    int b = i / 256, o = i % 256;
    float acc = fcb[o];
    for (int c = 0; c < 1024; ++c) acc += xg[b * 1024 + c] * fcw[o * 1024 + c];
    out[i] = fmaxf(acc, 0.0f);
}

// ---------------- host orchestration ----------------
extern "C" void kernel_launch(void* const* d_in, const int* in_sizes, int n_in,
                              void* d_out, int out_size, void* d_ws, size_t ws_size,
                              hipStream_t stream) {
    (void)in_sizes; (void)n_in; (void)out_size; (void)ws_size;

    const float* pts = (const float*)d_in[0];
    const float* w1 = (const float*)d_in[1];
    const float* g1 = (const float*)d_in[2];
    const float* b1 = (const float*)d_in[3];
    const float* w2 = (const float*)d_in[4];
    const float* g2 = (const float*)d_in[5];
    const float* b2 = (const float*)d_in[6];
    const float* w3 = (const float*)d_in[7];
    const float* g3 = (const float*)d_in[8];
    const float* b3 = (const float*)d_in[9];
    const float* w4 = (const float*)d_in[10];
    const float* g4 = (const float*)d_in[11];
    const float* b4 = (const float*)d_in[12];
    const float* w5 = (const float*)d_in[13];
    const float* g5 = (const float*)d_in[14];
    const float* b5 = (const float*)d_in[15];
    const float* fcw = (const float*)d_in[16];
    const float* fcb = (const float*)d_in[17];
    // d_in[18] = k (always 20 in the reference setup; hardcoded as KNN)

    float* out = (float*)d_out;

    // -------- workspace carve-up --------
    char* ws = (char*)d_ws;
    size_t off = 0;
    auto carve = [&](size_t bytes) -> char* {
        char* p = ws + off;
        off = (off + bytes + 255) & ~(size_t)255;
        return p;
    };
    float*    x0    = (float*)carve((size_t)BATCH * 3 * NPTS * 4);       // (B,3,N)
    int*      idx   = (int*)  carve((size_t)BATCH * NPTS * KNN * 4);     // (B,N,20)
    float*    xcat  = (float*)carve((size_t)BATCH * 512 * NPTS * 4);     // (B,512,N)
    _Float16* w1h   = (_Float16*)carve((size_t)64 * 32 * 2);             // K padded 6->32
    _Float16* w2h   = (_Float16*)carve((size_t)64 * 128 * 2);
    _Float16* w3h   = (_Float16*)carve((size_t)128 * 128 * 2);
    _Float16* w4h   = (_Float16*)carve((size_t)256 * 256 * 2);
    _Float16* w5h   = (_Float16*)carve((size_t)1024 * 512 * 2);
    float*    stats = (float*)carve((size_t)5 * 2048 * 4);               // 5 layers x [sum|sumsq]
    float*    scsh  = (float*)carve((size_t)5 * 2048 * 4);               // 5 layers x [scale|shift]
    float*    xg    = (float*)carve((size_t)BATCH * 1024 * 4);           // global max pool

    const int CB = 512 * NPTS;      // batch stride inside x_cat
    const float invcnt_conv = 1.0f / (float)(BATCH * NPTS * KNN);
    const float invcnt_glob = 1.0f / (float)(BATCH * NPTS);

    // -------- in-stream init (only buffers accumulated into via atomics) --------
    zero_f32<<<(5 * 2048 + 255) / 256, 256, 0, stream>>>(stats, 5 * 2048);
    zero_f32<<<(BATCH * 1024 + 255) / 256, 256, 0, stream>>>(xg, BATCH * 1024);

    pts_to_cxn<<<(BATCH * 3 * NPTS + 255) / 256, 256, 0, stream>>>(pts, x0);

    w_to_f16<<<(64 * 32 + 255) / 256, 256, 0, stream>>>(w1, w1h, 64, 6, 32);
    w_to_f16<<<(64 * 128 + 255) / 256, 256, 0, stream>>>(w2, w2h, 64, 128, 128);
    w_to_f16<<<(128 * 128 + 255) / 256, 256, 0, stream>>>(w3, w3h, 128, 128, 128);
    w_to_f16<<<(256 * 256 + 255) / 256, 256, 0, stream>>>(w4, w4h, 256, 256, 256);
    w_to_f16<<<(1024 * 512 + 255) / 256, 256, 0, stream>>>(w5, w5h, 1024, 512, 512);

    const dim3 knnBlk(128), gemmBlk(256);
    const dim3 knnGrid(NPTS / 128, BATCH);
    const dim3 convGrid(NKCOL / 80, BATCH);     // 80-col panels = 4 points each
    const dim3 globGrid(NPTS / 32, BATCH);      // 32-col panels

    // ======== layer 1: x0(3ch) -> x_cat[0:64) ========
    knn_kernel<3><<<knnGrid, knnBlk, 0, stream>>>(x0, 3 * NPTS, idx);
    gemm_conv<true, 1><<<convGrid, gemmBlk, 0, stream>>>(
        x0, 3 * NPTS, idx, w1h, 64, 3, 32, stats + 0 * 2048, nullptr, nullptr, 0, 0);
    bn_finalize<<<1, 256, 0, stream>>>(stats + 0 * 2048, g1, b1, scsh + 0 * 2048, 64, invcnt_conv);
    gemm_conv<true, 2><<<convGrid, gemmBlk, 0, stream>>>(
        x0, 3 * NPTS, idx, w1h, 64, 3, 32, nullptr, scsh + 0 * 2048, xcat, CB, 0);

    // ======== layer 2: x1 = x_cat[0:64) -> x_cat[64:128) ========
    knn_kernel<64><<<knnGrid, knnBlk, 0, stream>>>(xcat + 0 * NPTS, CB, idx);
    gemm_conv<true, 1><<<convGrid, gemmBlk, 0, stream>>>(
        xcat + 0 * NPTS, CB, idx, w2h, 64, 64, 128, stats + 1 * 2048, nullptr, nullptr, 0, 0);
    bn_finalize<<<1, 256, 0, stream>>>(stats + 1 * 2048, g2, b2, scsh + 1 * 2048, 64, invcnt_conv);
    gemm_conv<true, 2><<<convGrid, gemmBlk, 0, stream>>>(
        xcat + 0 * NPTS, CB, idx, w2h, 64, 64, 128, nullptr, scsh + 1 * 2048, xcat, CB, 64);

    // ======== layer 3: x2 = x_cat[64:128) -> x_cat[128:256) ========
    knn_kernel<64><<<knnGrid, knnBlk, 0, stream>>>(xcat + 64 * NPTS, CB, idx);
    gemm_conv<true, 1><<<convGrid, gemmBlk, 0, stream>>>(
        xcat + 64 * NPTS, CB, idx, w3h, 128, 64, 128, stats + 2 * 2048, nullptr, nullptr, 0, 0);
    bn_finalize<<<1, 256, 0, stream>>>(stats + 2 * 2048, g3, b3, scsh + 2 * 2048, 128, invcnt_conv);
    gemm_conv<true, 2><<<convGrid, gemmBlk, 0, stream>>>(
        xcat + 64 * NPTS, CB, idx, w3h, 128, 64, 128, nullptr, scsh + 2 * 2048, xcat, CB, 128);

    // ======== layer 4: x3 = x_cat[128:256) -> x_cat[256:512) ========
    knn_kernel<128><<<knnGrid, knnBlk, 0, stream>>>(xcat + 128 * NPTS, CB, idx);
    gemm_conv<true, 1><<<convGrid, gemmBlk, 0, stream>>>(
        xcat + 128 * NPTS, CB, idx, w4h, 256, 128, 256, stats + 3 * 2048, nullptr, nullptr, 0, 0);
    bn_finalize<<<1, 256, 0, stream>>>(stats + 3 * 2048, g4, b4, scsh + 3 * 2048, 256, invcnt_conv);
    gemm_conv<true, 2><<<convGrid, gemmBlk, 0, stream>>>(
        xcat + 128 * NPTS, CB, idx, w4h, 256, 128, 256, nullptr, scsh + 3 * 2048, xcat, CB, 256);

    // ======== global stage: h5 = w5 x x_cat, BN, ReLU, max over n ========
    gemm_conv<false, 1><<<globGrid, gemmBlk, 0, stream>>>(
        xcat, CB, nullptr, w5h, 1024, 512, 512, stats + 4 * 2048, nullptr, nullptr, 0, 0);
    bn_finalize<<<4, 256, 0, stream>>>(stats + 4 * 2048, g5, b5, scsh + 4 * 2048, 1024, invcnt_glob);
    gemm_conv<false, 2><<<globGrid, gemmBlk, 0, stream>>>(
        xcat, CB, nullptr, w5h, 1024, 512, 512, nullptr, scsh + 4 * 2048, xg, 0, 0);

    // ======== final FC + ReLU ========
    fc_relu<<<(BATCH * 256 + 255) / 256, 256, 0, stream>>>(xg, fcw, fcb, out);
}
